// NAT_76261439308315
// MI455X (gfx1250) — matrix operand order repaired and checked by reading, not compiled, not run on hardware
//
#include <hip/hip_runtime.h>
#include <hip/hip_bf16.h>
#include <math.h>

// ---------------------------------------------------------------------------
// NAT block for MI455X (gfx1250): all dense contractions via
// v_wmma_f32_16x16x32_bf16 (fp32 accumulate); LN/softmax in fp32.
// Staging uses GLOBAL_LOAD_ASYNC_TO_LDS_B128 (ASYNCcnt path).
// ---------------------------------------------------------------------------

typedef unsigned short u16;
typedef __attribute__((ext_vector_type(16))) __bf16         v16bf;
typedef __attribute__((ext_vector_type(16))) unsigned short v16u;
typedef __attribute__((ext_vector_type(8)))  float          v8f;
typedef int v4i __attribute__((vector_size(16)));

#define HW   9216   // 96*96 positions
#define WL   96     // spatial side
#define WP   98     // padded side
#define CDIM 64
#define C2   32
#define DH   8
#define HID  192

__device__ __forceinline__ u16 f2bf(float f) {
    union { float f; unsigned u; } c; c.f = f;
    unsigned u = c.u;
    unsigned r = u + 0x7FFFu + ((u >> 16) & 1u);   // round-to-nearest-even
    return (u16)(r >> 16);
}

enum { EPI_RELU = 1, EPI_ACC = 2, EPI_GELU = 4 };

// ------------------------- async LDS staging helpers -----------------------
#if defined(__has_builtin)
#if __has_builtin(__builtin_amdgcn_global_load_async_to_lds_b128)
#define HAVE_ASYNC_LDS 1
#endif
#endif
#ifndef HAVE_ASYNC_LDS
#define HAVE_ASYNC_LDS 0
#endif

typedef __attribute__((address_space(1))) v4i* gv4p;
typedef __attribute__((address_space(3))) v4i* lv4p;

__device__ __forceinline__ void stage16(const u16* __restrict__ g, u16* __restrict__ l) {
#if HAVE_ASYNC_LDS
    __builtin_amdgcn_global_load_async_to_lds_b128(
        (gv4p)(uintptr_t)g, (lv4p)(unsigned)(uintptr_t)l, 0, 0);
#else
    *(uint4*)l = *(const uint4*)g;
#endif
}

__device__ __forceinline__ void stage_wait() {
#if HAVE_ASYNC_LDS
#if __has_builtin(__builtin_amdgcn_s_wait_asynccnt)
    __builtin_amdgcn_s_wait_asynccnt(0);
#else
    asm volatile("s_wait_asynccnt 0" ::: "memory");
#endif
#endif
}

__device__ __forceinline__ v16bf ldfrag(const u16* p) {
    return __builtin_bit_cast(v16bf, *(const v16u*)p);
}

// ---------------------------------------------------------------------------
// Fragment-layout notes (CDNA5 ISA §7.12.2), bf16 16x16x32:
//  A (16xK): lane = r + 16*((kk>>3)&1), elem = (kk&7) + 8*(kk>>4)
//    -> elems 0..7 and 8..15 each cover 8 *consecutive* kk values.
//  B (Kx16): lane = n + 16*((kk>>3)&1), elem = (kk&7) + 8*(kk>>4)
//    -> with W stored transposed [N][K], elems 0..7 and 8..15 each cover 8 consecutive k.
//  C/D: VGPR r -> M = r + 8*(lane>>4), N = lane&15.
// So every LDS fragment is filled by contiguous 16-byte global runs.
// ---------------------------------------------------------------------------

// Common epilogue for one 16x16 accumulator.
__device__ __forceinline__ void store_tile(
    v8f a, int mb, int n, int N, const float* bias,
    float* Cf, u16* Cb, int flags, int padded)
{
    float bb = bias ? bias[n] : 0.0f;
    #pragma unroll
    for (int r = 0; r < 8; ++r) {
        float v = a[r] + bb;
        if (flags & EPI_RELU) v = fmaxf(v, 0.0f);
        if (flags & EPI_GELU) v = 0.5f * v * (1.0f + erff(v * 0.70710678118654752f));
        int m = mb + r;
        size_t idx;
        if (padded) {
            int xh = m / WL, yw = m - xh * WL;
            idx = (size_t)((xh + 1) * WP + yw + 1) * N + n;
        } else {
            idx = (size_t)m * N + n;
        }
        if (Cf) { if (flags & EPI_ACC) Cf[idx] += v; else Cf[idx] = v; }
        if (Cb) Cb[idx] = f2bf(v);
    }
}

// ---------------------------------------------------------------------------
// Linear GEMM: C[M,N] = epi(A[M,K] * Wt[N,K]^T + bias).  A bf16 row-major,
// Wt bf16 TRANSPOSED [N][K].  Tile 128(M) x 32(N), 256 thr = 8 waves,
// wave w owns M-rows [w*16, w*16+16) and both 16-col halves.
// ---------------------------------------------------------------------------
__global__ __launch_bounds__(256) void gemm_lin(
    const u16* __restrict__ A, const u16* __restrict__ Wt,
    const float* __restrict__ bias,
    float* __restrict__ Cf, u16* __restrict__ Cb,
    int M, int N, int K, int flags, int padded)
{
    __shared__ __align__(16) u16 As[8 * 512];
    __shared__ __align__(16) u16 Bs[2 * 512];
    const int tid = threadIdx.x, lane = tid & 31, wave = tid >> 5;
    const int m0 = blockIdx.x * 128, n0 = blockIdx.y * 32;

    // per-thread A transfer slots (2): one 16B run each
    const u16* ga[2]; u16* la[2];
    #pragma unroll
    for (int t = 0; t < 2; ++t) {
        int i  = tid + t * 256;
        int ms = i >> 6, ln = (i >> 1) & 31, hf = i & 1;
        int kkb = 8 * (ln >> 4) + 16 * hf;
        int m  = m0 + ms * 16 + (ln & 15);
        la[t] = &As[ms * 512 + ln * 16 + hf * 8];
        ga[t] = A + (size_t)m * K + kkb;
    }
    // B transfer slot (threads 0..127): one 16B run
    const u16* gb = nullptr; u16* lb = nullptr;
    {
        int j  = tid & 127;
        int ns = j >> 6, ln = (j >> 1) & 31, hf = j & 1;
        int kkb = 8 * (ln >> 4) + 16 * hf;
        int n  = n0 + ns * 16 + (ln & 15);
        lb = &Bs[ns * 512 + ln * 16 + hf * 8];
        gb = Wt + (size_t)n * K + kkb;
    }

    v8f acc0 = {}, acc1 = {};
    for (int k0 = 0; k0 < K; k0 += 32) {
        __syncthreads();
        stage16(ga[0] + k0, la[0]);
        stage16(ga[1] + k0, la[1]);
        if (tid < 128) stage16(gb + k0, lb);
        stage_wait();
        __syncthreads();
        v16bf av = ldfrag(&As[wave * 512 + lane * 16]);
        v16bf b0 = ldfrag(&Bs[lane * 16]);
        v16bf b1 = ldfrag(&Bs[512 + lane * 16]);
        acc0 = __builtin_amdgcn_wmma_f32_16x16x32_bf16(false, av, false, b0, (short)0, acc0, false, false);
        acc1 = __builtin_amdgcn_wmma_f32_16x16x32_bf16(false, av, false, b1, (short)0, acc1, false, false);
    }

    const int mb = m0 + wave * 16 + 8 * (lane >> 4);
    store_tile(acc0, mb, n0 + (lane & 15),      N, bias, Cf, Cb, flags, padded);
    store_tile(acc1, mb, n0 + 16 + (lane & 15), N, bias, Cf, Cb, flags, padded);
}

// ---------------------------------------------------------------------------
// Implicit-im2col conv GEMM.  A = zero-padded bf16 [98][98][Cin], K = 9*Cin.
// Within each 32-wide K chunk the 3x3 tap is uniform (k0 >> cinShift), so the
// gather address is branch-free: one scalar (dy,dx) + per-thread add.
// ---------------------------------------------------------------------------
__global__ __launch_bounds__(256) void gemm_conv(
    const u16* __restrict__ Ap, const u16* __restrict__ Wt,
    const float* __restrict__ bias,
    float* __restrict__ Cf, u16* __restrict__ Cb,
    int N, int K, int cinShift, int flags)
{
    __shared__ __align__(16) u16 As[8 * 512];
    __shared__ __align__(16) u16 Bs[2 * 512];
    const int tid = threadIdx.x, lane = tid & 31, wave = tid >> 5;
    const int m0 = blockIdx.x * 128, n0 = blockIdx.y * 32;
    const int Cin = 1 << cinShift;

    int pc[2], kkb[2]; u16* la[2];
    #pragma unroll
    for (int t = 0; t < 2; ++t) {
        int i  = tid + t * 256;
        int ms = i >> 6, ln = (i >> 1) & 31, hf = i & 1;
        kkb[t] = 8 * (ln >> 4) + 16 * hf;
        int m  = m0 + ms * 16 + (ln & 15);
        int xh = m / WL, yw = m - xh * WL;
        pc[t] = (xh + 1) * WP + (yw + 1);            // padded center
        la[t] = &As[ms * 512 + ln * 16 + hf * 8];
    }
    const u16* gb = nullptr; u16* lb = nullptr;
    {
        int j  = tid & 127;
        int ns = j >> 6, ln = (j >> 1) & 31, hf = j & 1;
        int kb = 8 * (ln >> 4) + 16 * hf;
        int n  = n0 + ns * 16 + (ln & 15);
        lb = &Bs[ns * 512 + ln * 16 + hf * 8];
        gb = Wt + (size_t)n * K + kb;
    }

    v8f acc0 = {}, acc1 = {};
    for (int k0 = 0; k0 < K; k0 += 32) {
        int tap  = k0 >> cinShift;                    // uniform per chunk
        int cinb = k0 & (Cin - 1);
        int ty   = tap / 3;
        int delta = (ty - 1) * WP + (tap - 3 * ty - 1);
        __syncthreads();
        stage16(Ap + (((size_t)(pc[0] + delta)) << cinShift) + cinb + kkb[0], la[0]);
        stage16(Ap + (((size_t)(pc[1] + delta)) << cinShift) + cinb + kkb[1], la[1]);
        if (tid < 128) stage16(gb + k0, lb);
        stage_wait();
        __syncthreads();
        v16bf av = ldfrag(&As[wave * 512 + lane * 16]);
        v16bf b0 = ldfrag(&Bs[lane * 16]);
        v16bf b1 = ldfrag(&Bs[512 + lane * 16]);
        acc0 = __builtin_amdgcn_wmma_f32_16x16x32_bf16(false, av, false, b0, (short)0, acc0, false, false);
        acc1 = __builtin_amdgcn_wmma_f32_16x16x32_bf16(false, av, false, b1, (short)0, acc1, false, false);
    }

    const int mb = m0 + wave * 16 + 8 * (lane >> 4);
    store_tile(acc0, mb, n0 + (lane & 15),      N, bias, Cf, Cb, flags, 0);
    store_tile(acc1, mb, n0 + 16 + (lane & 15), N, bias, Cf, Cb, flags, 0);
}

// ---------------------------------------------------------------------------
// LayerNorm over last dim (64); optionally writes into padded [98][98][64].
// ---------------------------------------------------------------------------
__global__ __launch_bounds__(256) void ln_to_bf16(
    const float* __restrict__ X, const float* __restrict__ g,
    const float* __restrict__ b, u16* __restrict__ out, int padded)
{
    int m = blockIdx.x * 256 + threadIdx.x;
    if (m >= HW) return;
    const float* row = X + (size_t)m * CDIM;
    float mean = 0.f;
    #pragma unroll 8
    for (int c = 0; c < CDIM; ++c) mean += row[c];
    mean *= (1.0f / CDIM);
    float var = 0.f;
    #pragma unroll 8
    for (int c = 0; c < CDIM; ++c) { float d = row[c] - mean; var += d * d; }
    var *= (1.0f / CDIM);
    float inv = rsqrtf(var + 1e-5f);
    size_t ob;
    if (padded) {
        int xh = m / WL, yw = m - xh * WL;
        ob = (size_t)((xh + 1) * WP + yw + 1) * CDIM;
    } else {
        ob = (size_t)m * CDIM;
    }
    #pragma unroll 8
    for (int c = 0; c < CDIM; ++c)
        out[ob + c] = f2bf((row[c] - mean) * inv * g[c] + b[c]);
}

// ---------------------------------------------------------------------------
// Neighborhood attention, online softmax. 1 thread = (position, head).
// qkv fp32 [HW][96]: q at h*8, k at 32+h*8, v at 64+h*8. out bf16 [HW][32].
// ---------------------------------------------------------------------------
__global__ __launch_bounds__(256) void nat_attn(
    const float* __restrict__ qkv, const float* __restrict__ rpb,
    u16* __restrict__ out, int Kw)
{
    int t   = blockIdx.x * 256 + threadIdx.x;
    int pos = t >> 2;
    int h   = t & 3;
    int xh  = pos / WL;
    int yw  = pos - xh * WL;
    int half = Kw >> 1;
    int stx = xh - half; stx = stx < 0 ? 0 : (stx > WL - Kw ? WL - Kw : stx);
    int sty = yw - half; sty = sty < 0 ? 0 : (sty > WL - Kw ? WL - Kw : sty);
    const int R = 2 * Kw - 1;
    const float scale = 0.35355339059327373f;   // 8^-0.5

    float q[DH];
    const float* qp = qkv + (size_t)pos * 96 + h * DH;
    #pragma unroll
    for (int d = 0; d < DH; ++d) q[d] = qp[d] * scale;

    float mx = -1e30f, sm = 0.f, o[DH] = {0, 0, 0, 0, 0, 0, 0, 0};
    const float* rb = rpb + (size_t)h * R * R;

    for (int i = 0; i < Kw; ++i) {
        int ii = stx + i;
        int bi = ii - xh + (Kw - 1);
        for (int j = 0; j < Kw; ++j) {
            int jj = sty + j;
            int bj = jj - yw + (Kw - 1);
            const float* kp = qkv + (size_t)(ii * WL + jj) * 96 + 32 + h * DH;
            float lg = rb[bi * R + bj];
            #pragma unroll
            for (int d = 0; d < DH; ++d) lg += q[d] * kp[d];
            float nm = fmaxf(mx, lg);
            float f  = __expf(mx - nm);
            float e  = __expf(lg - nm);
            sm = sm * f + e;
            const float* vp = kp + 32;
            #pragma unroll
            for (int d = 0; d < DH; ++d) o[d] = o[d] * f + e * vp[d];
            mx = nm;
        }
    }
    float inv = 1.0f / sm;
    #pragma unroll
    for (int d = 0; d < DH; ++d)
        out[(size_t)pos * C2 + h * DH + d] = f2bf(o[d] * inv);
}

// ---------------------------------------------------------------------------
// Param prep: fold BN into conv weights + TRANSPOSE to [N][K] bf16.
// (reference HWIO flat layout == [K][N] row-major, K = 9*Cin)
// ---------------------------------------------------------------------------
__global__ __launch_bounds__(256) void fold_conv_t(
    const float* __restrict__ w, const float* __restrict__ b,
    const float* __restrict__ g, const float* __restrict__ be,
    const float* __restrict__ mn, const float* __restrict__ vr,
    u16* __restrict__ wout, float* __restrict__ bout, int K, int N)
{
    int i = blockIdx.x * 256 + threadIdx.x;
    if (i < K * N) {
        int n = i / K, k = i - n * K;
        float s = g[n] * rsqrtf(vr[n] + 1e-5f);
        wout[i] = f2bf(w[(size_t)k * N + n] * s);
    }
    if (i < N) {
        float s = g[i] * rsqrtf(vr[i] + 1e-5f);
        bout[i] = (b[i] - mn[i]) * s + be[i];
    }
}

__global__ __launch_bounds__(256) void cvt_t(
    const float* __restrict__ w, u16* __restrict__ out, int K, int N)
{
    int i = blockIdx.x * 256 + threadIdx.x;
    if (i < K * N) {
        int n = i / K, k = i - n * K;
        out[i] = f2bf(w[(size_t)k * N + n]);
    }
}

__global__ __launch_bounds__(256) void copy_f32(
    const float* __restrict__ s, float* __restrict__ d, int n)
{
    int i = blockIdx.x * 256 + threadIdx.x;
    if (i < n) d[i] = s[i];
}

__global__ __launch_bounds__(256) void zero_u16(u16* __restrict__ p, int n)
{
    int i = blockIdx.x * 256 + threadIdx.x;
    if (i < n) p[i] = 0;
}

// ---------------------------------------------------------------------------
// Host-side orchestration
// ---------------------------------------------------------------------------

// d_in flattening (setup_inputs dict order, nested dicts in insertion order):
enum {
    IN_X = 0, IN_N1G, IN_N1B, IN_N2G, IN_N2B,
    IN_C1W, IN_C1B, IN_C1G, IN_C1BE, IN_C1M, IN_C1V,
    IN_C2W, IN_C2B, IN_C2G, IN_C2BE, IN_C2M, IN_C2V,
    IN_C22W, IN_C22B, IN_C22G, IN_C22BE, IN_C22M, IN_C22V,
    IN_C3W, IN_C3B, IN_C3G, IN_C3BE, IN_C3M, IN_C3V,
    IN_C4W, IN_C4B, IN_C4G, IN_C4BE, IN_C4M, IN_C4V,
    IN_C33W, IN_C33B, IN_C33G, IN_C33BE, IN_C33M, IN_C33V,
    IN_A15_QW, IN_A15_QB, IN_A15_PW, IN_A15_PB, IN_A15_RPB,
    IN_A7_QW,  IN_A7_QB,  IN_A7_PW,  IN_A7_PB,  IN_A7_RPB,
    IN_A3_QW,  IN_A3_QB,  IN_A3_PW,  IN_A3_PB,  IN_A3_RPB,
    IN_FC1W, IN_FC1B, IN_FC2W, IN_FC2B
};

extern "C" void kernel_launch(void* const* d_in, const int* in_sizes, int n_in,
                              void* d_out, int out_size, void* d_ws, size_t ws_size,
                              hipStream_t stream)
{
    (void)in_sizes; (void)n_in; (void)out_size; (void)ws_size;
    auto f32 = [&](int i) { return (const float*)d_in[i]; };

    // -------- workspace carve-out --------
    size_t off = 0;
    auto alloc = [&](size_t bytes) {
        void* p = (char*)d_ws + off;
        off += (bytes + 255) & ~(size_t)255;
        return p;
    };
    const int PAD64 = WP * WP * CDIM;   // padded xn
    const int PAD32 = WP * WP * C2;     // padded proj
    u16*   xnp     = (u16*)  alloc((size_t)PAD64 * 2);
    u16*   projp   = (u16*)  alloc((size_t)PAD32 * 2);
    u16*   y_bf    = (u16*)  alloc((size_t)HW * C2   * 2);
    float* qkv_f   = (float*)alloc((size_t)HW * 96   * 4);
    u16*   attn_bf = (u16*)  alloc((size_t)HW * C2   * 2);
    float* acc_f   = (float*)alloc((size_t)HW * CDIM * 4);
    u16*   xn2_bf  = (u16*)  alloc((size_t)HW * CDIM * 2);
    u16*   mlp_bf  = (u16*)  alloc((size_t)HW * HID  * 2);
    u16* cw1[3]; float* cb1[3];   // 64->32 convs: Wt [32][576]
    u16* cw2[3]; float* cb2[3];   // 32->64 convs: Wt [64][288]
    u16* qw[3];  u16* pw[3];
    for (int i = 0; i < 3; ++i) {
        cw1[i] = (u16*)alloc(576 * 32 * 2);  cb1[i] = (float*)alloc(32 * 4);
        cw2[i] = (u16*)alloc(288 * 64 * 2);  cb2[i] = (float*)alloc(64 * 4);
        qw[i]  = (u16*)alloc(32 * 96 * 2);   // Wt [96][32]
        pw[i]  = (u16*)alloc(32 * 32 * 2);   // Wt [32][32]
    }
    u16* fc1w = (u16*)alloc(64 * 192 * 2);   // Wt [192][64]
    u16* fc2w = (u16*)alloc(192 * 64 * 2);   // Wt [64][192]

    const int MT = HW / 128;  // 72 M-tiles

    // -------- 1. param prep + padded-buffer clears --------
    zero_u16<<<(PAD64 + 255) / 256, 256, 0, stream>>>(xnp, PAD64);
    zero_u16<<<(PAD32 + 255) / 256, 256, 0, stream>>>(projp, PAD32);
    const int cA[3] = { IN_C1W, IN_C2W, IN_C22W };
    const int cB[3] = { IN_C3W, IN_C4W, IN_C33W };
    for (int i = 0; i < 3; ++i) {
        fold_conv_t<<<(576 * 32 + 255) / 256, 256, 0, stream>>>(
            f32(cA[i] + 0), f32(cA[i] + 1), f32(cA[i] + 2), f32(cA[i] + 3),
            f32(cA[i] + 4), f32(cA[i] + 5), cw1[i], cb1[i], 576, 32);
        fold_conv_t<<<(288 * 64 + 255) / 256, 256, 0, stream>>>(
            f32(cB[i] + 0), f32(cB[i] + 1), f32(cB[i] + 2), f32(cB[i] + 3),
            f32(cB[i] + 4), f32(cB[i] + 5), cw2[i], cb2[i], 288, 64);
    }
    const int aQ[3] = { IN_A15_QW, IN_A7_QW, IN_A3_QW };
    for (int i = 0; i < 3; ++i) {
        cvt_t<<<(3072 + 255) / 256, 256, 0, stream>>>(f32(aQ[i] + 0), qw[i], 32, 96);
        cvt_t<<<(1024 + 255) / 256, 256, 0, stream>>>(f32(aQ[i] + 2), pw[i], 32, 32);
    }
    cvt_t<<<(12288 + 255) / 256, 256, 0, stream>>>(f32(IN_FC1W), fc1w, 64, 192);
    cvt_t<<<(12288 + 255) / 256, 256, 0, stream>>>(f32(IN_FC2W), fc2w, 192, 64);

    // -------- 2. LN1 (into padded buffer) + residual base --------
    ln_to_bf16<<<(HW + 255) / 256, 256, 0, stream>>>(
        f32(IN_X), f32(IN_N1G), f32(IN_N1B), xnp, 1);
    copy_f32<<<(HW * CDIM + 255) / 256, 256, 0, stream>>>(
        f32(IN_X), acc_f, HW * CDIM);

    // -------- 3. three NA branches --------
    const int kwin[3] = { 15, 7, 3 };
    for (int br = 0; br < 3; ++br) {
        // conv(64->32)+BN+ReLU (implicit im2col from padded xn), out bf16
        gemm_conv<<<dim3(MT, 1), 256, 0, stream>>>(
            xnp, cw1[br], cb1[br], nullptr, y_bf, 32, 576, 6, EPI_RELU);
        // qkv projection (K=32, N=96), out fp32
        gemm_lin<<<dim3(MT, 3), 256, 0, stream>>>(
            y_bf, qw[br], f32(aQ[br] + 1), qkv_f, nullptr,
            HW, 96, 32, 0, 0);
        // neighborhood attention
        nat_attn<<<(HW * 4) / 256, 256, 0, stream>>>(
            qkv_f, f32(aQ[br] + 4), attn_bf, kwin[br]);
        // output projection (K=32, N=32), out bf16 into padded buffer
        gemm_lin<<<dim3(MT, 1), 256, 0, stream>>>(
            attn_bf, pw[br], f32(aQ[br] + 3), nullptr, projp,
            HW, 32, 32, 0, /*padded=*/1);
        // conv(32->64)+BN+ReLU, accumulate into residual
        gemm_conv<<<dim3(MT, 2), 256, 0, stream>>>(
            projp, cw2[br], cb2[br], acc_f, nullptr, 64, 288, 5,
            EPI_RELU | EPI_ACC);
    }

    // -------- 4. LN2 + MLP + residual --------
    ln_to_bf16<<<(HW + 255) / 256, 256, 0, stream>>>(
        acc_f, f32(IN_N2G), f32(IN_N2B), xn2_bf, 0);
    gemm_lin<<<dim3(MT, HID / 32), 256, 0, stream>>>(
        xn2_bf, fc1w, f32(IN_FC1B), nullptr, mlp_bf,
        HW, HID, 64, EPI_GELU, 0);
    copy_f32<<<(HW * CDIM + 255) / 256, 256, 0, stream>>>(
        acc_f, (float*)d_out, HW * CDIM);
    gemm_lin<<<dim3(MT, 2), 256, 0, stream>>>(
        mlp_bf, fc2w, f32(IN_FC2B), (float*)d_out, nullptr,
        HW, CDIM, HID, EPI_ACC, 0);
}
